// PSCLoss_88708254531874
// MI455X (gfx1250) — matrix operand order repaired and checked by reading, not compiled, not run on hardware
//
#include <hip/hip_runtime.h>
#include <hip/hip_bf16.h>
#include <math.h>

typedef _Float16 half_t;
typedef __attribute__((ext_vector_type(16))) _Float16 v16h;
typedef __attribute__((ext_vector_type(8)))  float    v8f;
typedef __attribute__((ext_vector_type(4)))  unsigned int u32x4;
typedef int v4i __attribute__((vector_size(16)));   // type expected by async-LDS builtin

#define BATCH    4096
#define FEAT     512
#define NCLS     1000
#define NCLS_PAD 1024            // pad classes so 32 divides evenly (grid.y = 32)
#define KPROTO   10
#define NCOLS    (NCLS * KPROTO)      // 10000 real columns (dense, no K padding)
#define NTILES   (NCLS_PAD * KPROTO / 16)  // 640 N-tiles of 16 columns
#define TEMP_INV 10.0f

// GEMM macro-tile config
#define BM        64                   // batch rows per block (4 WMMA M-tiles)
#define BCLS      32                   // classes per block = 320 cols = 20 N-tiles
#define NT_BLK    20                   // N-tiles per block
#define PIECES    (4 + NT_BLK)         // 24 x 1KB pieces per K-step stage
#define STAGE_HALFS (PIECES * 512)     // 12288 halfs = 24 KB per buffer
#define STAGE_OPS 6                    // 16B async ops per thread (1536 total)

#if defined(__has_builtin)
#if __has_builtin(__builtin_amdgcn_global_load_async_to_lds_b128)
#define USE_ASYNC 1
#endif
#endif
#ifndef USE_ASYNC
#define USE_ASYNC 0
#endif

// ---------------------------------------------------------------------------
// Kernel 1: per-row inverse L2 norm (one wave32 per row)
// ---------------------------------------------------------------------------
__global__ __launch_bounds__(256) void k_norm(const float* __restrict__ feat,
                                              float* __restrict__ inv_norm) {
  int wave = (blockIdx.x * blockDim.x + threadIdx.x) >> 5;   // row id
  int lane = threadIdx.x & 31;
  if (wave >= BATCH) return;
  const float* row = feat + (size_t)wave * FEAT;
  float ss = 0.f;
#pragma unroll
  for (int i = 0; i < FEAT / 32; ++i) {
    float v = row[lane + 32 * i];
    ss += v * v;
  }
#pragma unroll
  for (int m = 16; m >= 1; m >>= 1) ss += __shfl_xor(ss, m, 32);
  if (lane == 0) inv_norm[wave] = 1.0f / fmaxf(sqrtf(ss), 1e-12f);
}

// ---------------------------------------------------------------------------
// Kernel 2: pack normalized feat into WMMA A-fragment order (f16).
// A tile = 16(M) x 32(K): lanes 0-15 hold K{0..7,16..23}, lanes 16-31 hold
// K{8..15,24..31}; 16 halfs per lane. Apack[(mt*16+kt)*512 + lane*16 + h]
// ---------------------------------------------------------------------------
__global__ __launch_bounds__(256) void k_packA(const float* __restrict__ feat,
                                               const float* __restrict__ inv_norm,
                                               half_t* __restrict__ Apack) {
  int idx  = blockIdx.x * 256 + threadIdx.x;      // 0 .. 4096*512-1
  int t    = idx >> 9;                            // tile id = mt*16+kt
  int w    = idx & 511;
  int lane = w >> 4, h = w & 15;
  int mt   = t >> 4, kt = t & 15;
  int m = mt * 16 + (lane & 15);
  int k = kt * 32 + ((h < 8) ? h : h + 8) + ((lane >= 16) ? 8 : 0);
  Apack[idx] = (half_t)(feat[(size_t)m * FEAT + k] * inv_norm[m]);
}

// ---------------------------------------------------------------------------
// Kernel 3: pack prototypes DENSELY into WMMA B-fragment order (f16).
// Flattened prototype row n = class*10 + protoIdx is contiguous in memory, so
// column n of B is just proto[n*512 + k]. Columns 10000..10239 are zero pad.
// B tile = 32(K) x 16(N): lanes 0-15 hold K{0..7,16..23} (N=lane), lanes 16-31 hold
// K{8..15,24..31} (N=lane-16). Bpack[(nt*16+kt)*512 + lane*16 + h]
// ---------------------------------------------------------------------------
__global__ __launch_bounds__(256) void k_packB(const float* __restrict__ proto,
                                               half_t* __restrict__ Bpack) {
  int idx = blockIdx.x * 256 + threadIdx.x;       // 0 .. 640*8192-1
  if (idx >= NTILES * 16 * 512) return;
  int nt   = idx >> 13;
  int w    = idx & 8191;
  int kt   = w >> 9;
  int ww   = w & 511;
  int lane = ww >> 4, h = ww & 15;
  int n = nt * 16 + (lane & 15);                  // global column 0..10239
  int k = kt * 32 + ((h < 8) ? h : h + 8) + ((lane >= 16) ? 8 : 0);
  float v = (n < NCOLS) ? proto[(size_t)n * FEAT + k] : 0.0f;
  Bpack[idx] = (half_t)v;
}

// ---------------------------------------------------------------------------
// Stage one K-step (32 deep) of the macro-tile into an LDS buffer:
// layout [A-tile0..3 | N-tile0..19], 512 halfs (1 KB) per piece.
// 6 x 16B async ops per thread; lanes of a wave cover a contiguous 512B chunk.
// ---------------------------------------------------------------------------
__device__ __forceinline__ void stage_tile(const half_t* __restrict__ Apack,
                                           const half_t* __restrict__ Bpack,
                                           half_t* dst, int mt0, int nt0,
                                           int kt, int tid) {
#pragma unroll
  for (int i = 0; i < STAGE_OPS; ++i) {
    int o = tid + 256 * i;          // 0..1535
    int p = o >> 6;                 // piece id 0..23
    int q = (o & 63) * 8;           // half offset within 1 KB piece
    int lofs = p * 512 + q;
    const half_t* g = (p < 4)
        ? (Apack + (((size_t)(mt0 + p) * 16 + kt) << 9) + q)
        : (Bpack + (((size_t)(nt0 + p - 4) * 16 + kt) << 9) + q);
#if USE_ASYNC
    __builtin_amdgcn_global_load_async_to_lds_b128(
        (v4i*)g, (v4i*)(dst + lofs), 0, 0);
#else
    *(u32x4*)(dst + lofs) = *(const u32x4*)g;
#endif
  }
}

// ---------------------------------------------------------------------------
// Kernel 4: WMMA GEMM (dense columns) + LDS-scoreboard max-over-10 epilogue.
// Block = 8 waves: wave w -> M-tile (w>>1), N-tiles (w&1)*10 .. +10.
// Per K-step per wave: 1 A frag (ds) reused across 10 WMMAs, 10 B frags (ds).
// ---------------------------------------------------------------------------
__global__ __launch_bounds__(256) void k_gemm(const half_t* __restrict__ Apack,
                                              const half_t* __restrict__ Bpack,
                                              float* __restrict__ logits) {
  __shared__ __align__(32) unsigned char smem[2 * STAGE_HALFS * sizeof(half_t)]; // 48 KB
  half_t* ldsbuf = (half_t*)smem;
  const int tid  = threadIdx.x;
  const int lane = tid & 31;
  const int wv   = tid >> 5;
  const int mt0  = blockIdx.x * (BM / 16);   // first of 4 M-tiles
  const int nt0  = blockIdx.y * NT_BLK;      // first of 20 N-tiles
  const int mi   = wv >> 1;                  // this wave's M-tile (0..3)
  const int nh   = wv & 1;                   // this wave's column half (10 N-tiles)

  v8f acc[10] = {};

  stage_tile(Apack, Bpack, ldsbuf, mt0, nt0, 0, tid);

  for (int kt = 0; kt < 16; ++kt) {
    half_t* cur = ldsbuf + (kt & 1) * STAGE_HALFS;
    if (kt < 15) {
      // prefetch next K-step into the other buffer, then wait only for the
      // current buffer's 6 ops (issued last iteration) -> DMA overlaps WMMA
      stage_tile(Apack, Bpack, ldsbuf + ((kt + 1) & 1) * STAGE_HALFS,
                 mt0, nt0, kt + 1, tid);
#if USE_ASYNC
      asm volatile("s_wait_asynccnt 0x6" ::: "memory");
#endif
    } else {
#if USE_ASYNC
      asm volatile("s_wait_asynccnt 0x0" ::: "memory");
#endif
    }
    __syncthreads();   // all waves' share of `cur` has landed in LDS

    v16h a = *(const v16h*)(cur + mi * 512 + lane * 16);
#pragma unroll
    for (int j = 0; j < 10; ++j) {
      v16h b = *(const v16h*)(cur + (4 + nh * 10 + j) * 512 + lane * 16);
      acc[j] = __builtin_amdgcn_wmma_f32_16x16x32_f16(
          false, a, false, b, (short)0, acc[j], false, false);
    }
    __syncthreads();   // done reading `cur` before it is re-staged
  }

  // Epilogue: max over groups of 10 columns via LDS scoreboard (aliases the
  // stage buffers; two phases of 160 columns keep it within 40 KB).
  // D layout: VGPR r -> lanes 0-15: (M=r, N=lane), lanes 16-31: (M=r+8, N=lane-16)
  float* sb = (float*)smem;                 // [64 rows][160 cols]
  const int colL  = lane & 15;
  const int rbase = mi * 16 + ((lane >= 16) ? 8 : 0);
#pragma unroll
  for (int ph = 0; ph < 2; ++ph) {
    __syncthreads();                        // previous smem users done
    if (nh == ph) {
#pragma unroll
      for (int j = 0; j < 10; ++j)
#pragma unroll
        for (int r = 0; r < 8; ++r)
          sb[(rbase + r) * 160 + j * 16 + colL] = acc[j][r];
    }
    __syncthreads();
    // 1024 outputs this phase: 64 rows x 16 local classes
#pragma unroll
    for (int i = 0; i < 4; ++i) {
      int o   = tid + 256 * i;
      int row = o >> 4, cl = o & 15;
      const float* s = sb + row * 160 + cl * 10;
      float v = s[0];
#pragma unroll
      for (int k = 1; k < 10; ++k) v = fmaxf(v, s[k]);
      int cls = blockIdx.y * BCLS + ph * 16 + cl;
      if (cls < NCLS)
        logits[(size_t)(blockIdx.x * BM + row) * NCLS + cls] = v;
    }
  }
}

// ---------------------------------------------------------------------------
// Kernel 5: per-row logsumexp over 1000 classes + positive-logit gather.
// ---------------------------------------------------------------------------
__global__ __launch_bounds__(128) void k_lse(const float* __restrict__ logits,
                                             const long long* __restrict__ label,
                                             float* __restrict__ rowvals) {
  __shared__ float red[128];
  const int row = blockIdx.x;
  const int tid = threadIdx.x;
  const float* lrow = logits + (size_t)row * NCLS;

  float m = -3.0e38f;
  for (int j = tid; j < NCLS; j += 128) m = fmaxf(m, lrow[j]);
  red[tid] = m; __syncthreads();
  for (int s = 64; s >= 1; s >>= 1) {
    if (tid < s) red[tid] = fmaxf(red[tid], red[tid + s]);
    __syncthreads();
  }
  m = red[0]; __syncthreads();

  float sum = 0.f;
  for (int j = tid; j < NCLS; j += 128) sum += __expf((lrow[j] - m) * TEMP_INV);
  red[tid] = sum; __syncthreads();
  for (int s = 64; s >= 1; s >>= 1) {
    if (tid < s) red[tid] += red[tid + s];
    __syncthreads();
  }
  if (tid == 0) {
    float lse = m * TEMP_INV + __logf(red[0]);
    int lab = (int)label[row];
    rowvals[row] = lrow[lab] * TEMP_INV - lse;
  }
}

// ---------------------------------------------------------------------------
// Kernel 6: final mean reduction, loss = -mean(rowvals)
// ---------------------------------------------------------------------------
__global__ __launch_bounds__(256) void k_final(const float* __restrict__ rowvals,
                                               float* __restrict__ out) {
  __shared__ float red[256];
  int tid = threadIdx.x;
  float s = 0.f;
  for (int i = tid; i < BATCH; i += 256) s += rowvals[i];
  red[tid] = s; __syncthreads();
  for (int k = 128; k >= 1; k >>= 1) {
    if (tid < k) red[tid] += red[tid + k];
    __syncthreads();
  }
  if (tid == 0) out[0] = -red[0] / (float)BATCH;
}

// ---------------------------------------------------------------------------
extern "C" void kernel_launch(void* const* d_in, const int* in_sizes, int n_in,
                              void* d_out, int out_size, void* d_ws, size_t ws_size,
                              hipStream_t stream) {
  const float*     feat  = (const float*)d_in[0];
  const long long* label = (const long long*)d_in[1];
  const float*     proto = (const float*)d_in[2];
  // d_in[3] (probs) and d_in[4] (sample_per_class) are unused by the reference.

  char* ws = (char*)d_ws;
  float*  inv_norm = (float*)ws;                         // 16 KiB
  half_t* Apack    = (half_t*)(ws + (64u  << 10));       // 4    MiB
  half_t* Bpack    = (half_t*)(ws + (8u   << 20));       // 10 MiB (640 N-tiles)
  float*  logits   = (float*)(ws + (20u  << 20));        // 15.63 MiB
  float*  rowvals  = (float*)(ws + (36u  << 20));        // 16 KiB
  float*  out      = (float*)d_out;

  k_norm <<<BATCH / 8, 256, 0, stream>>>(feat, inv_norm);
  k_packA<<<(BATCH * FEAT) / 256, 256, 0, stream>>>(feat, inv_norm, Apack);
  k_packB<<<(NTILES * 16 * 512) / 256, 256, 0, stream>>>(proto, Bpack);

  dim3 g(BATCH / BM, NTILES / NT_BLK);   // 64 x 32
  k_gemm <<<g, 256, 0, stream>>>(Apack, Bpack, logits);

  k_lse  <<<BATCH, 128, 0, stream>>>(logits, label, rowvals);
  k_final<<<1, 256, 0, stream>>>(rowvals, out);
}
